// MambaVote_59768764891602
// MI455X (gfx1250) — matrix operand order repaired and checked by reading, not compiled, not run on hardware
//
#include <hip/hip_runtime.h>

#define D_MODEL 256
#define D_STATE 16
#define D_CONV  4
#define D_INNER 512
#define DT_RANK 16
#define B_SZ    4
#define SEQ     4096
#define NROWS   (B_SZ * SEQ)      // 16384 rows through every GEMM
#define XPROJ_N 48                // DT_RANK + 2*D_STATE
#define CHUNK   64
#define NCH     (SEQ / CHUNK)     // 64 chunks -> 131072-way scan parallelism

typedef __attribute__((ext_vector_type(16))) __bf16 bf16x16;
typedef __attribute__((ext_vector_type(8)))  float  f32x8;

union Frag { uint4 q[2]; bf16x16 v; };

__device__ __forceinline__ unsigned short f2bf(float f) {
  unsigned int u = __float_as_uint(f);
  unsigned int r = u + 0x7FFFu + ((u >> 16) & 1u);   // round-to-nearest-even
  return (unsigned short)(r >> 16);
}
__device__ __forceinline__ float siluf(float x) { return x / (1.f + __expf(-x)); }
__device__ __forceinline__ float softplusf(float x) {
  return (x > 20.f) ? x : log1pf(__expf(x));
}

// ---------------------------------------------------------------------------
// Register-blocked bf16 WMMA GEMM: each wave computes a 64x64 block of
// C[M,N] = A[M,K] @ B[N,K]^T as 4x4 WMMA tiles -> 16 b128 loads : 16 wmma
// per K-step (vs 4:1 for a single tile). Layouts per CDNA5 ISA 7.12.2.
// ---------------------------------------------------------------------------
__global__ __launch_bounds__(256) void k_gemm_bf16_64x64(
    const unsigned short* __restrict__ A,   // [M,K] bf16 row-major
    const unsigned short* __restrict__ B,   // [N,K] bf16 row-major
    float* __restrict__ C, int M, int N, int K)
{
  int wid  = (blockIdx.x * blockDim.x + threadIdx.x) >> 5;
  int lane = threadIdx.x & 31;
  int ntile = N >> 6;
  int tm = wid / ntile;
  int tn = wid - tm * ntile;
  if (tm * 64 >= M) return;

  int lr = lane & 15;   // M index (A-frag) / N index (B-frag, C cols)
  int hi = lane >> 4;   // K-half selector

  const uint4* Aq = reinterpret_cast<const uint4*>(A);
  const uint4* Bq = reinterpret_cast<const uint4*>(B);

  long aq[4], bq[4];
  #pragma unroll
  for (int i = 0; i < 4; ++i) {
    aq[i] = ((long)(tm * 64 + i * 16 + lr) * (K >> 1) + hi * 4) >> 2;  // uint4 idx @k0=0
    bq[i] = ((long)(tn * 64 + i * 16 + lr) * (K >> 1) + hi * 4) >> 2;
  }

  f32x8 zero = {0.f, 0.f, 0.f, 0.f, 0.f, 0.f, 0.f, 0.f};
  f32x8 acc[4][4];
  #pragma unroll
  for (int i = 0; i < 4; ++i)
    #pragma unroll
    for (int j = 0; j < 4; ++j) acc[i][j] = zero;

  for (int k0 = 0; k0 < K; k0 += 32) {
    long ko = (long)(k0 >> 3);          // k0/2 dwords = k0/8 uint4s
    Frag af[4], bf[4];
    #pragma unroll
    for (int i = 0; i < 4; ++i) {
      af[i].q[0] = Aq[aq[i] + ko];      // A: K = hi*8..+7
      af[i].q[1] = Aq[aq[i] + ko + 2];  //    K = 16+hi*8..+7
      bf[i].q[0] = Bq[bq[i] + ko];      // B: K = hi*8..+7
      bf[i].q[1] = Bq[bq[i] + ko + 2];  //    K = 16+hi*8..+7
    }
    #pragma unroll
    for (int i = 0; i < 4; ++i)
      #pragma unroll
      for (int j = 0; j < 4; ++j)
        acc[i][j] = __builtin_amdgcn_wmma_f32_16x16x32_bf16(
            false, af[i].v, false, bf[j].v, (short)0, acc[i][j], false, false);
  }

  // C/D layout: VGPR r -> row r + 8*hi, col = lr
  #pragma unroll
  for (int i = 0; i < 4; ++i) {
    float* Crow = C + (long)(tm * 64 + i * 16 + hi * 8) * N + tn * 64 + lr;
    #pragma unroll
    for (int j = 0; j < 4; ++j)
      #pragma unroll
      for (int r = 0; r < 8; ++r) Crow[(long)r * N + j * 16] = acc[i][j][r];
  }
}

// Generic single-tile fallback (used for x_proj, N=48)
__global__ __launch_bounds__(256) void k_gemm_bf16(
    const unsigned short* __restrict__ A, const unsigned short* __restrict__ B,
    float* __restrict__ C, int M, int N, int K)
{
  int wid  = (blockIdx.x * blockDim.x + threadIdx.x) >> 5;
  int lane = threadIdx.x & 31;
  int ntile = N >> 4;
  int tm = wid / ntile;
  int tn = wid - tm * ntile;
  if (tm * 16 >= M) return;

  int lr = lane & 15;
  int hi = lane >> 4;

  long arowq = (long)(tm * 16 + lr) * (K >> 1);
  long browq = (long)(tn * 16 + lr) * (K >> 1);
  const uint4* Aq = reinterpret_cast<const uint4*>(A);
  const uint4* Bq = reinterpret_cast<const uint4*>(B);

  f32x8 acc = {0.f, 0.f, 0.f, 0.f, 0.f, 0.f, 0.f, 0.f};
  for (int k0 = 0; k0 < K; k0 += 32) {
    Frag a, b;
    long aq = (arowq + (long)(k0 >> 1) + hi * 4) >> 2;
    a.q[0] = Aq[aq];
    a.q[1] = Aq[aq + 2];
    long bq = (browq + (long)(k0 >> 1) + hi * 4) >> 2;
    b.q[0] = Bq[bq];
    b.q[1] = Bq[bq + 2];
    acc = __builtin_amdgcn_wmma_f32_16x16x32_bf16(false, a.v, false, b.v,
                                                  (short)0, acc, false, false);
  }
  float* Crow = C + (long)(tm * 16 + hi * 8) * N + tn * 16 + lr;
  #pragma unroll
  for (int i = 0; i < 8; ++i) Crow[(long)i * N] = acc[i];
}

// ---------------------------------------------------------------------------
// Elementwise staging kernels
// ---------------------------------------------------------------------------
__global__ void k_f32_to_bf16(const float* __restrict__ s, unsigned short* __restrict__ d, int n) {
  int i = blockIdx.x * blockDim.x + threadIdx.x;
  if (i < n) d[i] = f2bf(s[i]);
}

// stage g_data as bf16, optionally flipped along D_MODEL (backward branch)
__global__ void k_prep_x(const float* __restrict__ g, unsigned short* __restrict__ xb, int flip) {
  int i = blockIdx.x * blockDim.x + threadIdx.x;
  if (i >= NROWS * D_MODEL) return;
  int c = i & (D_MODEL - 1);
  int row = i >> 8;
  int cs = flip ? (D_MODEL - 1 - c) : c;
  xb[i] = f2bf(g[(long)row * D_MODEL + cs]);
}

// dt_w [D_INNER, 16] -> bf16 [D_INNER, 32] zero-padded so K is WMMA-sized
__global__ void k_pad_dtw(const float* __restrict__ w, unsigned short* __restrict__ wb) {
  int i = blockIdx.x * blockDim.x + threadIdx.x;
  if (i >= D_INNER * 32) return;
  int j = i & 31, d = i >> 5;
  wb[i] = (j < DT_RANK) ? f2bf(w[d * DT_RANK + j]) : (unsigned short)0;
}

__global__ void k_prep_A(const float* __restrict__ alog, float* __restrict__ A) {
  int i = blockIdx.x * blockDim.x + threadIdx.x;
  if (i < D_INNER * D_STATE) A[i] = -__expf(alog[i]);
}

// causal depthwise conv (width 4) + bias + silu; writes f32 (scan) + bf16 (x_proj GEMM)
__global__ void k_conv_silu(const float* __restrict__ xz, const float* __restrict__ cw,
                            const float* __restrict__ cb, float* __restrict__ xc,
                            unsigned short* __restrict__ xcb) {
  int i = blockIdx.x * blockDim.x + threadIdx.x;
  if (i >= NROWS * D_INNER) return;
  int d = i & (D_INNER - 1);
  int row = i >> 9;              // b*SEQ + l
  int l = row & (SEQ - 1);
  float s = cb[d];
  #pragma unroll
  for (int j = 0; j < D_CONV; ++j) {
    int ll = l - (D_CONV - 1) + j;
    if (ll >= 0)
      s += cw[d * D_CONV + j] * xz[(long)(row - (D_CONV - 1) + j) * (2 * D_INNER) + d];
  }
  float v = siluf(s);
  xc[i]  = v;
  xcb[i] = f2bf(v);
}

// x_dbl[:, :16] -> bf16 [NROWS, 32] zero-padded (dt_proj GEMM A-operand)
__global__ void k_extract_dt(const float* __restrict__ xdbl, unsigned short* __restrict__ dlb) {
  int i = blockIdx.x * blockDim.x + threadIdx.x;
  if (i >= NROWS * 32) return;
  int c = i & 31, row = i >> 5;
  dlb[i] = (c < DT_RANK) ? f2bf(xdbl[(long)row * XPROJ_N + c]) : (unsigned short)0;
}

// ---------------------------------------------------------------------------
// Chunked selective scan: h_l = exp(dt*A)*h_{l-1} + dt*B_l*u_l  (assoc. pairs)
// pass1: per-chunk (prod dA, h | h0=0); pass2: serial chunk combine (64 steps);
// pass3: replay chunk from true entering state, fuse D-skip + silu(z) gate.
// B/C rows of x_dbl are 16B-aligned -> float4 loads.
// ---------------------------------------------------------------------------
__global__ __launch_bounds__(256) void k_scan_pass1(
    const float* __restrict__ dtp, const float* __restrict__ dtb,
    const float* __restrict__ xdbl, const float* __restrict__ xc,
    const float* __restrict__ A, float* __restrict__ aprod, float* __restrict__ hpart)
{
  int tid = blockIdx.x * blockDim.x + threadIdx.x;       // ((b*NCH+ch)<<9)+d
  if (tid >= B_SZ * NCH * D_INNER) return;
  int d  = tid & (D_INNER - 1);
  int ch = (tid >> 9) & (NCH - 1);
  int b  = tid >> 15;
  float a[D_STATE], h[D_STATE], ap[D_STATE];
  #pragma unroll
  for (int n = 0; n < D_STATE; ++n) { a[n] = A[d * D_STATE + n]; h[n] = 0.f; ap[n] = 1.f; }
  float bias = dtb[d];
  int row0 = b * SEQ + ch * CHUNK;
  for (int t = 0; t < CHUNK; ++t) {
    int row = row0 + t;
    float dt = softplusf(dtp[(long)row * D_INNER + d] + bias);
    float u  = xc[(long)row * D_INNER + d];
    const float4* x4 = reinterpret_cast<const float4*>(xdbl + (long)row * XPROJ_N);
    union { float4 v[4]; float f[D_STATE]; } Bm;
    Bm.v[0] = x4[4]; Bm.v[1] = x4[5]; Bm.v[2] = x4[6]; Bm.v[3] = x4[7];
    float du = dt * u;
    #pragma unroll
    for (int n = 0; n < D_STATE; ++n) {
      float dA = __expf(dt * a[n]);
      h[n]  = dA * h[n] + du * Bm.f[n];
      ap[n] *= dA;
    }
  }
  #pragma unroll
  for (int n = 0; n < D_STATE; ++n) {
    aprod[(long)tid * D_STATE + n] = ap[n];
    hpart[(long)tid * D_STATE + n] = h[n];
  }
}

__global__ void k_scan_pass2(const float* __restrict__ aprod, const float* __restrict__ hpart,
                             float* __restrict__ hstart)
{
  int tid = blockIdx.x * blockDim.x + threadIdx.x;       // b*D_INNER + d
  if (tid >= B_SZ * D_INNER) return;
  int d = tid & (D_INNER - 1);
  int b = tid >> 9;
  float h[D_STATE];
  #pragma unroll
  for (int n = 0; n < D_STATE; ++n) h[n] = 0.f;
  for (int ch = 0; ch < NCH; ++ch) {
    long base = ((long)((b * NCH + ch) << 9) + d) * D_STATE;
    #pragma unroll
    for (int n = 0; n < D_STATE; ++n) {
      hstart[base + n] = h[n];                            // entering state of chunk
      h[n] = aprod[base + n] * h[n] + hpart[base + n];
    }
  }
}

__global__ __launch_bounds__(256) void k_scan_pass3(
    const float* __restrict__ dtp, const float* __restrict__ dtb,
    const float* __restrict__ xdbl, const float* __restrict__ xc,
    const float* __restrict__ A, const float* __restrict__ hstart,
    const float* __restrict__ Dp, const float* __restrict__ xz,
    unsigned short* __restrict__ yg)
{
  int tid = blockIdx.x * blockDim.x + threadIdx.x;
  if (tid >= B_SZ * NCH * D_INNER) return;
  int d  = tid & (D_INNER - 1);
  int ch = (tid >> 9) & (NCH - 1);
  int b  = tid >> 15;
  float a[D_STATE], h[D_STATE];
  #pragma unroll
  for (int n = 0; n < D_STATE; ++n) {
    a[n] = A[d * D_STATE + n];
    h[n] = hstart[(long)tid * D_STATE + n];
  }
  float bias = dtb[d];
  float Dd   = Dp[d];
  int row0 = b * SEQ + ch * CHUNK;
  for (int t = 0; t < CHUNK; ++t) {
    int row = row0 + t;
    float dt = softplusf(dtp[(long)row * D_INNER + d] + bias);
    float u  = xc[(long)row * D_INNER + d];
    const float4* x4 = reinterpret_cast<const float4*>(xdbl + (long)row * XPROJ_N);
    union { float4 v[4]; float f[D_STATE]; } Bm, Cm;
    Bm.v[0] = x4[4]; Bm.v[1] = x4[5];  Bm.v[2] = x4[6];  Bm.v[3] = x4[7];
    Cm.v[0] = x4[8]; Cm.v[1] = x4[9];  Cm.v[2] = x4[10]; Cm.v[3] = x4[11];
    float du = dt * u;
    float y = 0.f;
    #pragma unroll
    for (int n = 0; n < D_STATE; ++n) {
      float dA = __expf(dt * a[n]);
      h[n] = dA * h[n] + du * Bm.f[n];
      y += h[n] * Cm.f[n];
    }
    y += Dd * u;
    float zv = xz[(long)row * (2 * D_INNER) + D_INNER + d];
    yg[(long)row * D_INNER + d] = f2bf(y * siluf(zv));
  }
}

// out = fwd + flip_{D_MODEL}(bwd) + g
__global__ void k_combine(const float* __restrict__ g, const float* __restrict__ of,
                          const float* __restrict__ ob, float* __restrict__ out)
{
  int i = blockIdx.x * blockDim.x + threadIdx.x;
  if (i >= NROWS * D_MODEL) return;
  int c = i & (D_MODEL - 1);
  int row = i >> 8;
  out[i] = of[i] + ob[(long)row * D_MODEL + (D_MODEL - 1 - c)] + g[i];
}

// ---------------------------------------------------------------------------
extern "C" void kernel_launch(void* const* d_in, const int* in_sizes, int n_in,
                              void* d_out, int out_size, void* d_ws, size_t ws_size,
                              hipStream_t stream) {
  const float* g = (const float*)d_in[0];
  float* out = (float*)d_out;

  char* p = (char*)d_ws;
  auto alloc = [&](size_t bytes) -> void* {
    void* r = (void*)p;
    p += (bytes + 255) & ~(size_t)255;
    return r;
  };
  float* xz            = (float*)alloc(sizeof(float) * NROWS * 2 * D_INNER);
  float* xc            = (float*)alloc(sizeof(float) * NROWS * D_INNER);
  unsigned short* xcb  = (unsigned short*)alloc(2ul * NROWS * D_INNER);
  float* xdbl          = (float*)alloc(sizeof(float) * NROWS * XPROJ_N);
  unsigned short* dlb  = (unsigned short*)alloc(2ul * NROWS * 32);
  float* dtp           = (float*)alloc(sizeof(float) * NROWS * D_INNER);
  unsigned short* ygb  = (unsigned short*)alloc(2ul * NROWS * D_INNER);
  unsigned short* xb   = (unsigned short*)alloc(2ul * NROWS * D_MODEL);
  unsigned short* wb_in  = (unsigned short*)alloc(2ul * 2 * D_INNER * D_MODEL);
  unsigned short* wb_xp  = (unsigned short*)alloc(2ul * XPROJ_N * D_INNER);
  unsigned short* wb_dt  = (unsigned short*)alloc(2ul * D_INNER * 32);
  unsigned short* wb_out = (unsigned short*)alloc(2ul * D_MODEL * D_INNER);
  float* Abuf   = (float*)alloc(sizeof(float) * D_INNER * D_STATE);
  float* aprod  = (float*)alloc(sizeof(float) * B_SZ * NCH * D_INNER * D_STATE);
  float* hpart  = (float*)alloc(sizeof(float) * B_SZ * NCH * D_INNER * D_STATE);
  float* hstart = (float*)alloc(sizeof(float) * B_SZ * NCH * D_INNER * D_STATE);
  float* dout0  = (float*)alloc(sizeof(float) * NROWS * D_MODEL);
  float* dout1  = (float*)alloc(sizeof(float) * NROWS * D_MODEL);
  float* dirout[2] = { dout0, dout1 };

  const int TB = 256;
  auto blocks = [](long n) { return (int)((n + 255) / 256); };
  auto gemm = [&](const unsigned short* A, const unsigned short* B, float* C,
                  int M, int N, int K) {
    if ((M & 63) == 0 && (N & 63) == 0) {
      int waves = (M / 64) * (N / 64);
      k_gemm_bf16_64x64<<<(waves + 7) / 8, TB, 0, stream>>>(A, B, C, M, N, K);
    } else {
      int waves = (M / 16) * (N / 16);
      k_gemm_bf16<<<(waves + 7) / 8, TB, 0, stream>>>(A, B, C, M, N, K);
    }
  };

  for (int dir = 0; dir < 2; ++dir) {
    const float* in_w   = (const float*)d_in[1 + 9 * dir];
    const float* conv_w = (const float*)d_in[2 + 9 * dir];
    const float* conv_b = (const float*)d_in[3 + 9 * dir];
    const float* xp_w   = (const float*)d_in[4 + 9 * dir];
    const float* dt_w   = (const float*)d_in[5 + 9 * dir];
    const float* dt_b   = (const float*)d_in[6 + 9 * dir];
    const float* A_log  = (const float*)d_in[7 + 9 * dir];
    const float* D_p    = (const float*)d_in[8 + 9 * dir];
    const float* out_w  = (const float*)d_in[9 + 9 * dir];

    // Stage weights/activations to bf16
    k_f32_to_bf16<<<blocks(2 * D_INNER * D_MODEL), TB, 0, stream>>>(in_w, wb_in, 2 * D_INNER * D_MODEL);
    k_f32_to_bf16<<<blocks(XPROJ_N * D_INNER), TB, 0, stream>>>(xp_w, wb_xp, XPROJ_N * D_INNER);
    k_pad_dtw<<<blocks(D_INNER * 32), TB, 0, stream>>>(dt_w, wb_dt);
    k_f32_to_bf16<<<blocks(D_MODEL * D_INNER), TB, 0, stream>>>(out_w, wb_out, D_MODEL * D_INNER);
    k_prep_A<<<blocks(D_INNER * D_STATE), TB, 0, stream>>>(A_log, Abuf);
    k_prep_x<<<blocks((long)NROWS * D_MODEL), TB, 0, stream>>>(g, xb, dir);

    // in_proj: xz[16384,1024] = x @ in_w^T
    gemm(xb, wb_in, xz, NROWS, 2 * D_INNER, D_MODEL);

    // causal depthwise conv + silu
    k_conv_silu<<<blocks((long)NROWS * D_INNER), TB, 0, stream>>>(xz, conv_w, conv_b, xc, xcb);

    // x_proj: x_dbl[16384,48] = xc @ xproj_w^T
    gemm(xcb, wb_xp, xdbl, NROWS, XPROJ_N, D_INNER);

    // dt_proj: dtp[16384,512] = dt_low(pad32) @ dt_w(pad32)^T
    k_extract_dt<<<blocks((long)NROWS * 32), TB, 0, stream>>>(xdbl, dlb);
    gemm(dlb, wb_dt, dtp, NROWS, D_INNER, 32);

    // chunked selective scan (+ fused softplus, D-skip, silu(z) gate -> bf16)
    k_scan_pass1<<<blocks((long)B_SZ * NCH * D_INNER), TB, 0, stream>>>(
        dtp, dt_b, xdbl, xc, Abuf, aprod, hpart);
    k_scan_pass2<<<blocks((long)B_SZ * D_INNER), TB, 0, stream>>>(aprod, hpart, hstart);
    k_scan_pass3<<<blocks((long)B_SZ * NCH * D_INNER), TB, 0, stream>>>(
        dtp, dt_b, xdbl, xc, Abuf, hstart, D_p, xz, ygb);

    // out_proj: dirout[16384,256] = y_gated @ out_w^T
    gemm(ygb, wb_out, dirout[dir], NROWS, D_MODEL, D_INNER);
  }

  k_combine<<<blocks((long)NROWS * D_MODEL), TB, 0, stream>>>(g, dirout[0], dirout[1], out);
}
